// OneLevelMoE_47081431499390
// MI455X (gfx1250) — matrix operand order repaired and checked by reading, not compiled, not run on hardware
//
#include <hip/hip_runtime.h>
#include <hip/hip_bf16.h>

typedef __attribute__((ext_vector_type(16))) _Float16     v16h;
typedef __attribute__((ext_vector_type(8)))  float        v8f;
typedef __attribute__((ext_vector_type(4)))  int          v4i;
typedef __attribute__((ext_vector_type(8)))  int          v8i;
typedef __attribute__((ext_vector_type(4)))  unsigned int v4ui;

#define B_TOK 16384
#define D_IN  256
#define H_DIM 1024
#define E_EXP 16

// packed layout (in _Float16 units):
//   17 matrices (gate_w1, exp_w1[0..15]), each K=256 x N=1024, fragment order:
//     stage s = mat*64 + nt  ->  packed + s*4096   (8KB per stage)
//   then gate_w2 (K=1024 x N=16) fragments at W2_OFF_H.
#define W1_MAT_H (D_IN * H_DIM)          // 262144 halves (512 KB) per matrix
#define W1_ALL_H (17 * W1_MAT_H)
#define W2_OFF_H W1_ALL_H
#define N_STAGES (17 * 64)               // 1088

#ifndef __has_builtin
#define __has_builtin(x) 0
#endif
#if __has_builtin(__builtin_amdgcn_tensor_load_to_lds)
#define USE_TDM 1
#else
#define USE_TDM 0
#endif
#if __has_builtin(__builtin_amdgcn_global_load_async_to_lds_b128)
#define USE_ASYNC_LDS 1
#else
#define USE_ASYNC_LDS 0
#endif

#define GPTR(p) ((__attribute__((address_space(1))) v4i*)(p))
#define LPTR(p) ((__attribute__((address_space(3))) v4i*)(p))

__device__ __forceinline__ void wait_async_zero() {
#if __has_builtin(__builtin_amdgcn_s_wait_asynccnt)
  __builtin_amdgcn_s_wait_asynccnt(0);
#else
  asm volatile("s_wait_asynccnt 0x0" ::: "memory");
#endif
}

__device__ __forceinline__ void wait_tensor_zero() {
#if __has_builtin(__builtin_amdgcn_s_wait_tensorcnt)
  __builtin_amdgcn_s_wait_tensorcnt(0);
#else
  asm volatile("s_wait_tensorcnt 0x0" ::: "memory");
#endif
}

// ---------------------------------------------------------------------------
// Pack f32 [K,N] weights into f16 WMMA B-fragment order.
// B fragment (32x16, f16): lane<16  -> N=lane,    K = ks*32 + 0..15
//                          lane>=16 -> N=lane-16, K = ks*32 + 16..31
// ---------------------------------------------------------------------------
__global__ __launch_bounds__(256) void pack_w1_kernel(
    const float* __restrict__ gate_w1, const float* __restrict__ exp_w1,
    _Float16* __restrict__ packed) {
  int tid = blockIdx.x * 256 + threadIdx.x;          // 17*64*8*32 = 278528
  if (tid >= 17 * 64 * 8 * 32) return;
  int mat  = tid >> 14;
  int rem  = tid & 16383;
  int nt   = rem >> 8;
  int rem2 = rem & 255;
  int ks   = rem2 >> 5;
  int lane = rem2 & 31;
  const float* src = (mat == 0) ? gate_w1
                                : (exp_w1 + (size_t)(mat - 1) * D_IN * H_DIM);
  int col = nt * 16 + (lane & 15);
  int kb  = ks * 32 + ((lane & 16) ? 16 : 0);
  _Float16* dst = packed + (size_t)mat * W1_MAT_H + nt * 4096 + ks * 512 + lane * 16;
#pragma unroll
  for (int j = 0; j < 16; ++j)
    dst[j] = (_Float16)src[(size_t)(kb + j) * H_DIM + col];
}

__global__ __launch_bounds__(256) void pack_w2_kernel(
    const float* __restrict__ gate_w2, _Float16* __restrict__ packed) {
  int tid = blockIdx.x * 256 + threadIdx.x;          // 32 ksteps * 32 lanes
  if (tid >= 1024) return;
  int ks   = tid >> 5;
  int lane = tid & 31;
  int col  = lane & 15;
  int kb   = ks * 32 + ((lane & 16) ? 16 : 0);
  _Float16* dst = packed + W2_OFF_H + ks * 512 + lane * 16;
#pragma unroll
  for (int j = 0; j < 16; ++j)
    dst[j] = (_Float16)gate_w2[(kb + j) * E_EXP + col];
}

// ---------------------------------------------------------------------------
// Fused MoE compute: 8 wave32s x 16 tokens = 128 tokens per workgroup.
// Single software-pipelined loop over all 17*64 weight stages; each next 8KB
// weight stage is DMA'd global->LDS by the Tensor Data Mover (one descriptor,
// issued by wave 0) while all waves run WMMAs; 1 barrier per stage.
// ---------------------------------------------------------------------------
__global__ __launch_bounds__(256) void moe_compute_kernel(
    const float* __restrict__ x,
    const float* __restrict__ gate_b1, const float* __restrict__ gate_b2,
    const float* __restrict__ exp_b1,  const float* __restrict__ exp_w2,
    const float* __restrict__ exp_b2,
    const _Float16* __restrict__ packed,
    float* __restrict__ out, float* __restrict__ gw_out) {
  __shared__ alignas(16) _Float16 smem_w[2][4096];    // 2 x 8 KB weight stages
  __shared__ alignas(16) _Float16 smem_g[8][512];     // per-wave 16x32 transpose

  const int tid  = threadIdx.x;
  const int lane = tid & 31;
  const int wave = tid >> 5;                           // 0..7 (wave32)
  const int tb   = blockIdx.x * 128 + wave * 16;       // wave token base
  const int colL = lane & 15;
  const int hi   = lane >> 4;                          // lane half

  // ---- A fragments: x[tb..tb+15][0..255] -> 8 x (16x32 f16), k = ks*32 + 16*hi + j (as B) ----
  v16h a[8];
  {
    const float* xr = x + (size_t)(tb + colL) * D_IN;
    const int o = hi ? 16 : 0;
#pragma unroll
    for (int ks = 0; ks < 8; ++ks) {
      v16h f;
#pragma unroll
      for (int j = 0; j < 8; ++j) {
        f[j]     = (_Float16)xr[ks * 32 + o + j];
        f[j + 8] = (_Float16)xr[ks * 32 + o + 8 + j];
      }
      a[ks] = f;
    }
  }

  // ---- prologue: stage 0 into buffer 0 ----
  {
    const uint4* s4 = (const uint4*)packed;
    uint4*       d4 = (uint4*)smem_w[0];
    d4[tid]       = s4[tid];
    d4[tid + 256] = s4[tid + 256];
  }
  __syncthreads();

  const v16h* w2f = (const v16h*)(packed + W2_OFF_H);
  v8f   logits = {};
  float gw[8]      = {0.f, 0.f, 0.f, 0.f, 0.f, 0.f, 0.f, 0.f};
  float gwe[8]     = {0.f, 0.f, 0.f, 0.f, 0.f, 0.f, 0.f, 0.f};
  float sum[8]     = {0.f, 0.f, 0.f, 0.f, 0.f, 0.f, 0.f, 0.f};
  float out_acc[8] = {0.f, 0.f, 0.f, 0.f, 0.f, 0.f, 0.f, 0.f};

  for (int s = 0; s < N_STAGES; ++s) {
    const int  cur = s & 1, nxt = cur ^ 1;
    const int  mat = s >> 6, nt = s & 63;
    const bool have_next = (s + 1) < N_STAGES;
    const _Float16* nsrc = packed + (size_t)(s + 1) * 4096;

    // ---- launch next stage's copy (overlaps with the WMMAs below) ----
#if USE_TDM
    if (have_next && wave == 0) {
      // Tensor DMA descriptor: 1-row tile of 1024 x 8-byte elements (8 KB).
      unsigned lds_addr = (unsigned)(uintptr_t)(
          (__attribute__((address_space(3))) void*)&smem_w[nxt][0]);
      unsigned long long ga = (unsigned long long)(uintptr_t)nsrc;
      v4ui g0;
      g0[0] = 1u;                                       // count=1, user D#
      g0[1] = lds_addr;                                 // LDS byte address
      g0[2] = (unsigned)ga;                             // global addr [31:0]
      g0[3] = (unsigned)((ga >> 32) & 0x1FFFFFFu)       // global addr [56:32]
              | (2u << 30);                             // type = 2 ("image")
      v8i g1;
      g1[0] = (int)(3u << 16);          // wg_mask=0, data_size=3 (8B), no flags
      g1[1] = (int)(1024u << 16);       // tensor_dim0[15:0]=1024 (hi16)
      g1[2] = (int)(1u << 16);          // tensor_dim0 hi=0 | tensor_dim1=1
      g1[3] = (int)(1024u << 16);       // tensor_dim1 hi=0 | tile_dim0=1024
      g1[4] = 1;                        // tile_dim1=1, tile_dim2=0
      g1[5] = 1024;                     // tensor_dim0_stride[31:0]
      g1[6] = 0;                        // stride hi | tensor_dim1_stride lo
      g1[7] = 0;                        // tensor_dim1_stride hi
      v4i gz4 = {0, 0, 0, 0};
      v8i gz8 = {0, 0, 0, 0, 0, 0, 0, 0};
      __builtin_amdgcn_tensor_load_to_lds(g0, g1, gz4, gz4, gz8, 0);
    }
#elif USE_ASYNC_LDS
    if (have_next) {
      __builtin_amdgcn_global_load_async_to_lds_b128(
          GPTR(nsrc + tid * 8), LPTR(&smem_w[nxt][tid * 8]), 0, 0);
      __builtin_amdgcn_global_load_async_to_lds_b128(
          GPTR(nsrc + 2048 + tid * 8), LPTR(&smem_w[nxt][2048 + tid * 8]), 0, 0);
    }
#else
    uint4 r0 = {}, r1 = {};
    if (have_next) {
      r0 = ((const uint4*)nsrc)[tid];
      r1 = ((const uint4*)nsrc)[tid + 256];
    }
#endif

    // ---- 16x16 tile GEMM: 8 k-steps, B fragment pipelined 1 deep ----
    v8f acc = {};
    {
      const v16h* bw = (const v16h*)smem_w[cur];
      v16h bcur = bw[lane];
#pragma unroll
      for (int ks = 0; ks < 8; ++ks) {
        v16h bnext = bcur;
        if (ks < 7) bnext = bw[(ks + 1) * 32 + lane];   // issue ahead of WMMA
        acc = __builtin_amdgcn_wmma_f32_16x16x32_f16(
            false, a[ks], false, bcur, (short)0, acc, false, false);
        bcur = bnext;
      }
    }

    if (mat == 0) {
      // ---------------- gate layer 1 + fused layer 2 ----------------
      const float b1 = gate_b1[nt * 16 + colL];
      _Float16* g = smem_g[wave];
#pragma unroll
      for (int r = 0; r < 8; ++r) {
        float v = fmaxf(acc[r] + b1, 0.f);
        g[(r + hi * 8) * 32 + (nt & 1) * 16 + colL] = (_Float16)v;
      }
      if (nt & 1) {            // a 16x32 chunk of relu(g) is ready -> logits step
        const int o = hi ? 16 : 0;
        v16h at;
#pragma unroll
        for (int j = 0; j < 8; ++j) {
          at[j]     = g[colL * 32 + o + j];
          at[j + 8] = g[colL * 32 + o + 8 + j];
        }
        logits = __builtin_amdgcn_wmma_f32_16x16x32_f16(
            false, at, false, w2f[(nt >> 1) * 32 + lane], (short)0, logits,
            false, false);
      }
      if (nt == 63) {          // softmax over experts (16 lanes per half)
        const float b2 = gate_b2[colL];
#pragma unroll
        for (int r = 0; r < 8; ++r) {
          float l = logits[r] + b2;
          float mx = l;
          for (int msk = 8; msk >= 1; msk >>= 1)
            mx = fmaxf(mx, __shfl_xor(mx, msk, 32));
          float ex = __expf(l - mx);
          float ssum = ex;
          for (int msk = 8; msk >= 1; msk >>= 1)
            ssum += __shfl_xor(ssum, msk, 32);
          gw[r] = ex / ssum;
        }
#pragma unroll
        for (int r = 0; r < 8; ++r)
          gw_out[(size_t)(tb + r + hi * 8) * E_EXP + colL] = gw[r];
      }
    } else {
      // ---------------------- expert mat-1 --------------------------
      const int e = mat - 1;
      if (nt == 0) {
#pragma unroll
        for (int r = 0; r < 8; ++r) {
          gwe[r] = __shfl(gw[r], (hi << 4) + e, 32);   // broadcast gw[.][e]
          sum[r] = 0.f;
        }
      }
      const float b1 = exp_b1[e * H_DIM + nt * 16 + colL];
      const float w2 = exp_w2[e * H_DIM + nt * 16 + colL];
#pragma unroll
      for (int r = 0; r < 8; ++r)
        sum[r] = fmaf(fmaxf(acc[r] + b1, 0.f), w2, sum[r]);
      if (nt == 63) {          // reduce columns across the 16-lane half
#pragma unroll
        for (int r = 0; r < 8; ++r) {
          float v = sum[r];
          for (int msk = 8; msk >= 1; msk >>= 1) v += __shfl_xor(v, msk, 32);
          out_acc[r] = fmaf(v + exp_b2[e], gwe[r], out_acc[r]);
        }
      }
    }

    // ---- drain the copy engine, single barrier per stage ----
#if USE_TDM
    if (wave == 0) wait_tensor_zero();
#elif USE_ASYNC_LDS
    wait_async_zero();
#else
    if (have_next) {
      ((uint4*)smem_w[nxt])[tid]       = r0;
      ((uint4*)smem_w[nxt])[tid + 256] = r1;
    }
#endif
    __syncthreads();
  }

  if (colL == 0) {
#pragma unroll
    for (int r = 0; r < 8; ++r) out[tb + r + hi * 8] = out_acc[r];
  }
}

// ---------------------------------------------------------------------------
extern "C" void kernel_launch(void* const* d_in, const int* in_sizes, int n_in,
                              void* d_out, int out_size, void* d_ws, size_t ws_size,
                              hipStream_t stream) {
  const float* x       = (const float*)d_in[0];
  const float* gate_w1 = (const float*)d_in[1];
  const float* gate_b1 = (const float*)d_in[2];
  const float* gate_w2 = (const float*)d_in[3];
  const float* gate_b2 = (const float*)d_in[4];
  const float* exp_w1  = (const float*)d_in[5];
  const float* exp_b1  = (const float*)d_in[6];
  const float* exp_w2  = (const float*)d_in[7];
  const float* exp_b2  = (const float*)d_in[8];

  float* out    = (float*)d_out;          // [B]   (weighted expert output)
  float* gw_out = out + B_TOK;            // [B,E] (gate weights)
  _Float16* packed = (_Float16*)d_ws;     // ~9 MB scratch

  pack_w1_kernel<<<1088, 256, 0, stream>>>(gate_w1, exp_w1, packed);
  pack_w2_kernel<<<4, 256, 0, stream>>>(gate_w2, packed);
  moe_compute_kernel<<<B_TOK / 128, 256, 0, stream>>>(
      x, gate_b1, gate_b2, exp_b1, exp_w2, exp_b2, packed, out, gw_out);
}
